// VNetwork_Mix_67456756351050
// MI455X (gfx1250) — matrix operand order repaired and checked by reading, not compiled, not run on hardware
//
#include <hip/hip_runtime.h>

// ---------------------------------------------------------------------------
// Fused VNetwork-Mix for MI455X (gfx1250, wave32, WMMA f32_16x16x32_f16).
// B=131072 rows, D=128, N=4 teachers. One wave owns a 16-row tile end-to-end.
// ---------------------------------------------------------------------------

typedef __attribute__((ext_vector_type(16))) _Float16 v16h;
typedef __attribute__((ext_vector_type(8)))  _Float16 v8h;
typedef __attribute__((ext_vector_type(8)))  float    v8f;

#define WAVES   3
#define THREADS (WAVES * 32)
#define NROWS   131072
#define NTILES  (NROWS / 16)   // 8192
#define NBLOCKS ((NTILES + WAVES - 1) / WAVES)  // 2731

static __device__ __forceinline__ v16h cat16(v8h lo, v8h hi) {
  return __builtin_shufflevector(lo, hi, 0,1,2,3,4,5,6,7,8,9,10,11,12,13,14,15);
}

// ---------------------------------------------------------------------------
// One layer: 5 matrices (4 teachers + student), K -> N, all-wave shared weight
// stage in LDS, per-wave activations.  LAST==true: layer 3 (teacher acts dead,
// student output feeds the 64->1 head accumulator in registers).
// ---------------------------------------------------------------------------
template <int K, int N, bool LAST>
static __device__ __forceinline__ void run_layer(
    const float* const* Wp,          // 5 weight matrices, row-major (K x N) f32
    const float* const* Bp,          // 5 bias vectors (N) f32
    const float* P, float m,
    bool valid, int lane,
    const _Float16* inBase, int inMatStride,   // f16 acts [16][K] per mat (stride 0 => shared x)
    _Float16* outBase,                          // f16 acts [5][16][N] (unused if LAST)
    _Float16* wStage,                           // shared B-fragment staging area
    float* partial4, const float* sW4)          // head accumulator (LAST only)
{
  constexpr int KT = K / 32;   // k-tiles of 32
  constexpr int NT = N / 16;   // n-tiles of 16

  float sum_acc[NT][8];
  #pragma unroll
  for (int nt = 0; nt < NT; ++nt)
    #pragma unroll
    for (int r = 0; r < 8; ++r) sum_acc[nt][r] = 0.0f;

  for (int mt = 0; mt < 5; ++mt) {
    // ---- stage W[mt] into LDS, pre-swizzled into WMMA B-fragment order ----
    __syncthreads();   // previous stage fully consumed
    {
      const float* Wsrc = Wp[mt];
      const int nslots = KT * NT * 32;        // one 32B slot per (kt,nt,lane)
      for (int s = (int)threadIdx.x; s < nslots; s += THREADS) {
        const int l   = s & 31;
        const int ntc = (s >> 5) % NT;
        const int ktc = (s >> 5) / NT;
        const int n   = ntc * 16 + (l & 15);
        const int kb  = ktc * 32 + ((l & 16) ? 8 : 0);   // K: kb..kb+7, kb+16..kb+23
        _Float16* dst = wStage + (size_t)s * 16;
        const float* src = Wsrc + (size_t)kb * N + n;
        #pragma unroll
        for (int h = 0; h < 16; ++h) dst[h] = (_Float16)src[(size_t)(h < 8 ? h : h + 8) * N];
      }
    }
    __syncthreads();   // stage visible to all waves

    if (valid) {
      // ---- A fragments for this matrix's input acts (ISA 16-bit A layout) ----
      const _Float16* act = inBase + (size_t)mt * inMatStride;
      v16h a[KT];
      #pragma unroll
      for (int kt = 0; kt < KT; ++kt) {
        const int rrow = lane & 15;
        const int c0 = kt * 32 + ((lane & 16) ? 8 : 0);
        v8h lo = *(const v8h*)(act + (size_t)rrow * K + c0);
        v8h hi = *(const v8h*)(act + (size_t)rrow * K + c0 + 16);
        a[kt] = cat16(lo, hi);
      }

      const float* bias = Bp[mt];
      #pragma unroll
      for (int nt = 0; nt < NT; ++nt) {
        v8f acc = {};
        #pragma unroll
        for (int kt = 0; kt < KT; ++kt) {
          const _Float16* wp = wStage + (size_t)(((kt * NT + nt) << 5) + lane) * 16;
          v8h blo = *(const v8h*)wp;
          v8h bhi = *(const v8h*)(wp + 8);
          v16h b = cat16(blo, bhi);
          acc = __builtin_amdgcn_wmma_f32_16x16x32_f16(
              false, a[kt], false, b, (short)0, acc, false, false);
        }
        const float bv = bias[nt * 16 + (lane & 15)];

        if (mt < 4) {                 // teacher: accumulate mix, store relu act
          const float p = P[mt];
          #pragma unroll
          for (int r = 0; r < 8; ++r) {
            const float pre = acc[r] + bv;
            sum_acc[nt][r] = __builtin_fmaf(p, pre, sum_acc[nt][r]);
            if (!LAST) {
              const int mrow = r + ((lane & 16) ? 8 : 0);
              outBase[(size_t)mt * 16 * N + (size_t)mrow * N + nt * 16 + (lane & 15)] =
                  (_Float16)fmaxf(pre, 0.0f);
            }
          }
        } else {                      // student: blend with teacher mix, relu
          #pragma unroll
          for (int r = 0; r < 8; ++r) {
            const float pre = acc[r] + bv;
            const float h = fmaxf(pre * (1.0f - m) + sum_acc[nt][r] * m, 0.0f);
            if (!LAST) {
              const int mrow = r + ((lane & 16) ? 8 : 0);
              outBase[(size_t)4 * 16 * N + (size_t)mrow * N + nt * 16 + (lane & 15)] =
                  (_Float16)h;
            } else {                  // fuse 64->1 head: out += h3 * sW4[col]
              const float w4 = sW4[nt * 16 + (lane & 15)];
              partial4[r] = __builtin_fmaf(h, w4, partial4[r]);
            }
          }
        }
      }
    }
  }
}

__global__ __launch_bounds__(THREADS) void vmix_fused_kernel(
    const float* __restrict__ x, const float* __restrict__ mixf,
    const float* __restrict__ P,
    const float* __restrict__ sW1, const float* __restrict__ sb1,
    const float* __restrict__ sW2, const float* __restrict__ sb2,
    const float* __restrict__ sW3, const float* __restrict__ sb3,
    const float* __restrict__ sW4, const float* __restrict__ sb4,
    const float* __restrict__ tW1, const float* __restrict__ tb1,
    const float* __restrict__ tW2, const float* __restrict__ tb2,
    const float* __restrict__ tW3, const float* __restrict__ tb3,
    float* __restrict__ out)
{
  __shared__ __attribute__((aligned(16))) _Float16 wStage[16384];          // 32 KB
  __shared__ __attribute__((aligned(16))) _Float16 actbuf[WAVES][2][5*16*128]; // 120 KB
  __shared__ __attribute__((aligned(16))) float    red[WAVES][16][16];     // 3 KB

  const int tid  = (int)threadIdx.x;
  const int wave = tid >> 5;
  const int lane = tid & 31;
  const int tile = (int)blockIdx.x * WAVES + wave;
  const bool valid = tile < NTILES;     // wave-uniform => EXEC all-1s for WMMA
  const int row0 = tile * 16;

  const float m = mixf[0];

  // ---- stage this wave's 16x128 x-tile into LDS as f16 (row-major) ----
  if (valid) {
    const float* xs = x + (size_t)row0 * 128;
    _Float16* xd = &actbuf[wave][0][0];
    for (int i = lane; i < 16 * 128; i += 32) xd[i] = (_Float16)xs[i];
  }

  float partial4[8];
  #pragma unroll
  for (int r = 0; r < 8; ++r) partial4[r] = 0.0f;

  // ---- layer 1: x(128) -> 128, all 5 matrices read the shared x tile ----
  {
    const float* W[5] = { tW1, tW1 + 16384, tW1 + 32768, tW1 + 49152, sW1 };
    const float* Bv[5] = { tb1, tb1 + 128, tb1 + 256, tb1 + 384, sb1 };
    run_layer<128, 128, false>(W, Bv, P, m, valid, lane,
                               &actbuf[wave][0][0], /*inMatStride=*/0,
                               &actbuf[wave][1][0], wStage, partial4, sW4);
  }
  // ---- layer 2: 128 -> 128 ----
  {
    const float* W[5] = { tW2, tW2 + 16384, tW2 + 32768, tW2 + 49152, sW2 };
    const float* Bv[5] = { tb2, tb2 + 128, tb2 + 256, tb2 + 384, sb2 };
    run_layer<128, 128, false>(W, Bv, P, m, valid, lane,
                               &actbuf[wave][1][0], /*inMatStride=*/16 * 128,
                               &actbuf[wave][0][0], wStage, partial4, sW4);
  }
  // ---- layer 3: 128 -> 64, fused with 64->1 head ----
  {
    const float* W[5] = { tW3, tW3 + 8192, tW3 + 16384, tW3 + 24576, sW3 };
    const float* Bv[5] = { tb3, tb3 + 64, tb3 + 128, tb3 + 192, sb3 };
    run_layer<128, 64, true>(W, Bv, P, m, valid, lane,
                             &actbuf[wave][0][0], /*inMatStride=*/16 * 128,
                             &actbuf[wave][1][0], wStage, partial4, sW4);
  }

  // ---- reduce the head partials across lanes (D-layout: M = r + 8*(lane/16)) ----
  __syncthreads();
  if (valid) {
    #pragma unroll
    for (int r = 0; r < 8; ++r)
      red[wave][r + ((lane & 16) ? 8 : 0)][lane & 15] = partial4[r];
  }
  __syncthreads();
  if (valid && lane < 16) {
    float s = sb4[0];
    #pragma unroll
    for (int j = 0; j < 16; ++j) s += red[wave][lane][j];
    out[row0 + lane] = s;
  }
}

extern "C" void kernel_launch(void* const* d_in, const int* in_sizes, int n_in,
                              void* d_out, int out_size, void* d_ws, size_t ws_size,
                              hipStream_t stream) {
  (void)in_sizes; (void)n_in; (void)d_ws; (void)ws_size; (void)out_size;
  const float* x    = (const float*)d_in[0];
  const float* mixf = (const float*)d_in[1];
  const float* P    = (const float*)d_in[2];
  const float* sW1  = (const float*)d_in[3];
  const float* sb1  = (const float*)d_in[4];
  const float* sW2  = (const float*)d_in[5];
  const float* sb2  = (const float*)d_in[6];
  const float* sW3  = (const float*)d_in[7];
  const float* sb3  = (const float*)d_in[8];
  const float* sW4  = (const float*)d_in[9];
  const float* sb4  = (const float*)d_in[10];
  const float* tW1  = (const float*)d_in[11];
  const float* tb1  = (const float*)d_in[12];
  const float* tW2  = (const float*)d_in[13];
  const float* tb2  = (const float*)d_in[14];
  const float* tW3  = (const float*)d_in[15];
  const float* tb3  = (const float*)d_in[16];
  float* out = (float*)d_out;

  vmix_fused_kernel<<<NBLOCKS, THREADS, 0, stream>>>(
      x, mixf, P, sW1, sb1, sW2, sb2, sW3, sb3, sW4, sb4,
      tW1, tb1, tW2, tb2, tW3, tb3, out);
}
